// ScaledDotProductAttention_45148696216954
// MI455X (gfx1250) — matrix operand order repaired and checked by reading, not compiled, not run on hardware
//
#include <hip/hip_runtime.h>
#include <hip/hip_bf16.h>

typedef __attribute__((ext_vector_type(16))) __bf16 v16bf;
typedef __attribute__((ext_vector_type(8)))  __bf16 v8bf;
typedef __attribute__((ext_vector_type(4)))  __bf16 v4bf;
typedef __attribute__((ext_vector_type(8)))  float  v8f;
typedef __attribute__((ext_vector_type(4)))  float  v4f;

#define S_LEN   1024
#define D_LEN   128
#define BM      64     // query rows per workgroup (4 row-tiles of 16)
#define BN      64     // keys per tile (16 per key-slice wave)
#define NW      16     // waves per workgroup
#define NTHREADS 512
#define QSTR    136    // bf16 elems per Q/K LDS row (128 + 8 pad; 16B-aligned rows)
#define VTS     72     // bf16 elems per Vt LDS row (64 + 8 pad)
#define PSTR    72     // bf16 elems per P LDS row
#define PLS     68     // f32 elems per prev LDS row (64 + 4 pad; 16B-aligned rows)

__device__ __forceinline__ v8f wmma_bf16(v16bf a, v16bf b, v8f c) {
  // D = A(16x32 bf16) * B(32x16 bf16) + C(16x16 f32)
  return __builtin_amdgcn_wmma_f32_16x16x32_bf16(
      /*neg_a=*/false, a, /*neg_b=*/false, b,
      /*c_mod=*/(short)0, c, /*reuse_a=*/false, /*reuse_b=*/false);
}

__device__ __forceinline__ v16bf cat16(v8bf lo, v8bf hi) {
  return __builtin_shufflevector(lo, hi, 0, 1, 2, 3, 4, 5, 6, 7,
                                         8, 9, 10, 11, 12, 13, 14, 15);
}

// Async DMA: 16 bytes per lane, global -> LDS, tracked by ASYNCcnt (ISA 08 §4).
__device__ __forceinline__ void async_load_b128(unsigned lds_off, unsigned goff,
                                                unsigned long long gbase) {
  asm volatile("global_load_async_to_lds_b128 %0, %1, %2 offset:0"
               :: "v"(lds_off), "v"(goff), "s"(gbase)
               : "memory");
}

__device__ __forceinline__ void wait_async0() {
  asm volatile("s_wait_asynccnt 0x0" ::: "memory");
}

__global__ __launch_bounds__(NTHREADS)
void attn_fwd_kernel(const float* __restrict__ Q, const float* __restrict__ K,
                     const float* __restrict__ V, const float* __restrict__ P,
                     float* __restrict__ Out, float* __restrict__ Logits) {
  __shared__ __bf16 Qs[BM * QSTR];
  __shared__ __bf16 Ks[BN * QSTR];
  __shared__ __bf16 Vt[D_LEN * VTS];     // transposed: [d][key]
  __shared__ __bf16 Ps[BM * PSTR];
  __shared__ float  Pls[2 * BM * PLS];   // double-buffered prev tiles (async DMA)
  __shared__ float  wm[NW * 16];
  __shared__ float  ws[NW * 16];

  const int tid  = threadIdx.x;
  const int w    = tid >> 5;       // wave 0..15
  const int lane = tid & 31;
  const int l15  = lane & 15;
  const int hi   = lane >> 4;      // half-wave select
  const int wq   = w >> 2;         // row-tile 0..3 (16 rows each)
  const int wn   = w & 3;          // key-slice (QK) / D-slice (PV) 0..3
  const int rq   = wq * 16;        // row base within block
  const int bh   = blockIdx.y;     // 0..63  (B*H)
  const int q0   = blockIdx.x * BM;

  const float inv_t = 0.088388347648318447f;  // 1/sqrt(128)

  const float* qptr = Q + (size_t)bh * S_LEN * D_LEN;
  const float* kptr = K + (size_t)bh * S_LEN * D_LEN;
  const float* vptr = V + (size_t)bh * S_LEN * D_LEN;
  const float* pptr = P + (size_t)bh * S_LEN * S_LEN;
  float* optr = Out    + (size_t)bh * S_LEN * D_LEN;
  float* lptr = Logits + (size_t)bh * S_LEN * S_LEN;
  const unsigned long long pbase = (unsigned long long)(size_t)pptr;

  // ---- issue async DMA of the first prev tile (64x64 f32) into buffer 0 ----
  {
#pragma unroll
    for (int it = 0; it < 2; ++it) {
      const int idx = it * NTHREADS + tid;   // 0..1023 chunks of 16B
      const int row = idx >> 4;              // 0..63
      const int c16 = idx & 15;              // 4-float chunk within row
      const unsigned goff =
          (unsigned)((((q0 + row) * S_LEN) + c16 * 4) * sizeof(float));
      const unsigned loff = (unsigned)(size_t)(&Pls[row * PLS + c16 * 4]);
      async_load_b128(loff, goff, pbase);
    }
  }

  // ---- stage Q tile (64 x 128) as bf16 in LDS ----
  for (int idx = tid; idx < BM * (D_LEN / 4); idx += NTHREADS) {
    const int r = idx >> 5, c4 = idx & 31;
    const v4f qv = *(const v4f*)(qptr + (size_t)(q0 + r) * D_LEN + c4 * 4);
    v4bf qb;
    qb[0] = (__bf16)qv[0]; qb[1] = (__bf16)qv[1];
    qb[2] = (__bf16)qv[2]; qb[3] = (__bf16)qv[3];
    *(v4bf*)(&Qs[r * QSTR + c4 * 4]) = qb;
  }

  v8f o0 = {}, o1 = {};
  float Mreg[8], Lreg[8];
#pragma unroll
  for (int r = 0; r < 8; ++r) { Mreg[r] = -__builtin_inff(); Lreg[r] = 0.f; }

  const int ntiles = (q0 + BM - 1) / BN + 1;   // causal: tiles up to the diagonal

  for (int j = 0; j < ntiles; ++j) {
    wait_async0();     // prev tile j (issued last iteration / prologue) resident
    __syncthreads();   // all waves' DMA visible; prior-iter LDS readers done
    const int kbase = j * BN;

    // ---- stage K tile (64 x 128) and V tile transposed (128 x 64), bf16 ----
    for (int idx = tid; idx < BN * (D_LEN / 4); idx += NTHREADS) {
      const int r = idx >> 5, c4 = idx & 31;
      const v4f kv = *(const v4f*)(kptr + (size_t)(kbase + r) * D_LEN + c4 * 4);
      v4bf kb;
      kb[0] = (__bf16)kv[0]; kb[1] = (__bf16)kv[1];
      kb[2] = (__bf16)kv[2]; kb[3] = (__bf16)kv[3];
      *(v4bf*)(&Ks[r * QSTR + c4 * 4]) = kb;
      const v4f vv = *(const v4f*)(vptr + (size_t)(kbase + r) * D_LEN + c4 * 4);
#pragma unroll
      for (int i = 0; i < 4; ++i) Vt[(c4 * 4 + i) * VTS + r] = (__bf16)vv[i];
    }
    __syncthreads();

    // ---- overlap: async DMA of next prev tile + prefetch next K/V tile ----
    if (j + 1 < ntiles) {
      float* nbuf = &Pls[((j + 1) & 1) * BM * PLS];
#pragma unroll
      for (int it = 0; it < 2; ++it) {
        const int idx = it * NTHREADS + tid;
        const int row = idx >> 4;
        const int c16 = idx & 15;
        const unsigned goff = (unsigned)(
            (((q0 + row) * S_LEN) + (kbase + BN) + c16 * 4) * sizeof(float));
        const unsigned loff = (unsigned)(size_t)(&nbuf[row * PLS + c16 * 4]);
        async_load_b128(loff, goff, pbase);
      }
      const char* nk = (const char*)(kptr + (size_t)(kbase + BN) * D_LEN);
      const char* nv = (const char*)(vptr + (size_t)(kbase + BN) * D_LEN);
      __builtin_prefetch(nk + (size_t)tid * 128, 0, 0);  // 512 thr * 128B = 64KB
      __builtin_prefetch(nv + (size_t)tid * 128, 0, 0);
    }

    // ---- S = Q * K^T : this wave's 16x16 tile ----
    v8f st = {};
#pragma unroll
    for (int kc = 0; kc < D_LEN; kc += 32) {
      const __bf16* qrow = &Qs[(rq + l15) * QSTR + kc];
      const v16bf A = cat16(*(const v8bf*)(qrow + 16 * hi),
                            *(const v8bf*)(qrow + 16 * hi + 8));
      const __bf16* krow = &Ks[(wn * 16 + l15) * QSTR + kc + 16 * hi];
      const v16bf Bm = cat16(*(const v8bf*)(krow), *(const v8bf*)(krow + 8));
      st = wmma_bf16(A, Bm, st);
    }

    // ---- scale, +prev (from async-DMA'd LDS tile), mask, emit logits (NT) ----
    const float* pls = &Pls[(j & 1) * BM * PLS];
    const int gcol = kbase + wn * 16 + l15;
#pragma unroll
    for (int r = 0; r < 8; ++r) {
      const int row16 = r + 8 * hi;
      const int grow  = q0 + rq + row16;
      float val = st[r] * inv_t + pls[(rq + row16) * PLS + wn * 16 + l15];
      if (gcol > grow) val = -__builtin_inff();
      st[r] = val;
      __builtin_nontemporal_store(val, lptr + (size_t)grow * S_LEN + gcol);
    }

    // ---- per-row max over this wave's 16 key columns ----
#pragma unroll
    for (int r = 0; r < 8; ++r) {
      float m = st[r];
      m = fmaxf(m, __shfl_xor(m, 1, 16));
      m = fmaxf(m, __shfl_xor(m, 2, 16));
      m = fmaxf(m, __shfl_xor(m, 4, 16));
      m = fmaxf(m, __shfl_xor(m, 8, 16));
      if (l15 == 0) wm[w * 16 + r + 8 * hi] = m;
    }
    __syncthreads();

    // ---- combine across the 4 key-slice waves of this row-tile ----
    float alpha[8];
#pragma unroll
    for (int r = 0; r < 8; ++r) {
      const int row16 = r + 8 * hi;
      const int sbase = wq * 4 * 16 + row16;
      const float tm = fmaxf(fmaxf(wm[sbase + 0 * 16], wm[sbase + 1 * 16]),
                             fmaxf(wm[sbase + 2 * 16], wm[sbase + 3 * 16]));
      const float newM = fmaxf(Mreg[r], tm);
      alpha[r] = __expf(Mreg[r] - newM);
      Mreg[r] = newM;
      const float p = __expf(st[r] - newM);
      Ps[(rq + row16) * PSTR + wn * 16 + l15] = (__bf16)p;
      float s = p;
      s += __shfl_xor(s, 1, 16);
      s += __shfl_xor(s, 2, 16);
      s += __shfl_xor(s, 4, 16);
      s += __shfl_xor(s, 8, 16);
      if (l15 == 0) ws[w * 16 + row16] = s;
    }
    __syncthreads();

#pragma unroll
    for (int r = 0; r < 8; ++r) {
      const int sbase = wq * 4 * 16 + r + 8 * hi;
      const float ts = ws[sbase + 0 * 16] + ws[sbase + 1 * 16] +
                       ws[sbase + 2 * 16] + ws[sbase + 3 * 16];
      Lreg[r] = Lreg[r] * alpha[r] + ts;
      o0[r] *= alpha[r];
      o1[r] *= alpha[r];
    }

    // ---- O += P(16x64) * V(64 x 32-slice of D) ----
    const int vc = wn * 32 + l15;
#pragma unroll
    for (int kc = 0; kc < BN; kc += 32) {
      const __bf16* prow = &Ps[(rq + l15) * PSTR + kc];
      const v16bf A = cat16(*(const v8bf*)(prow + 16 * hi),
                            *(const v8bf*)(prow + 16 * hi + 8));
      const __bf16* v0 = &Vt[vc * VTS + kc + 16 * hi];
      const v16bf B0 = cat16(*(const v8bf*)(v0), *(const v8bf*)(v0 + 8));
      o0 = wmma_bf16(A, B0, o0);
      const __bf16* v1 = &Vt[(vc + 16) * VTS + kc + 16 * hi];
      const v16bf B1 = cat16(*(const v8bf*)(v1), *(const v8bf*)(v1 + 8));
      o1 = wmma_bf16(A, B1, o1);
    }
  }

  // ---- normalize and write O ----
  const int oc = wn * 32 + l15;
#pragma unroll
  for (int r = 0; r < 8; ++r) {
    const int grow = q0 + rq + r + 8 * hi;
    const float invL = 1.0f / Lreg[r];
    optr[(size_t)grow * D_LEN + oc]      = o0[r] * invL;
    optr[(size_t)grow * D_LEN + oc + 16] = o1[r] * invL;
  }

  // ---- fill fully-masked logit columns with -inf (streaming NT stores) ----
  const int c0 = ntiles * BN;
  for (int r = 0; r < BM; ++r) {
    const size_t base = (size_t)(q0 + r) * S_LEN;
    for (int c = c0 + tid; c < S_LEN; c += NTHREADS)
      __builtin_nontemporal_store(-__builtin_inff(), lptr + base + c);
  }
}

extern "C" void kernel_launch(void* const* d_in, const int* in_sizes, int n_in,
                              void* d_out, int out_size, void* d_ws, size_t ws_size,
                              hipStream_t stream) {
  (void)in_sizes; (void)n_in; (void)out_size; (void)d_ws; (void)ws_size;
  const float* q    = (const float*)d_in[0];
  const float* k    = (const float*)d_in[1];
  const float* v    = (const float*)d_in[2];
  // d_in[3] is the boolean causal mask; computed analytically in-kernel.
  const float* prev = (const float*)d_in[4];

  float* out    = (float*)d_out;                               // [B,H,S,D]
  float* logits = out + (size_t)4 * 16 * 1024 * 128;           // [B,H,S,S]

  dim3 grid(S_LEN / BM, 4 * 16);   // 16 row-blocks x 64 (b,h)
  attn_fwd_kernel<<<grid, NTHREADS, 0, stream>>>(q, k, v, prev, out, logits);
}
